// MetaLayer_30227979829536
// MI455X (gfx1250) — matrix operand order repaired and checked by reading, not compiled, not run on hardware
//
#include <hip/hip_runtime.h>

// ---------------------------------------------------------------------------
// MetaLayer GraphNet step for MI455X (gfx1250, wave32, WMMA bf16 16x16x32).
// All four stage GEMMs run on v_wmma_f32_16x16x32_bf16 with f32 accumulate.
// ---------------------------------------------------------------------------

typedef __attribute__((ext_vector_type(16))) __bf16          v16bf;
typedef __attribute__((ext_vector_type(8)))  float           v8f;
typedef __attribute__((ext_vector_type(16))) unsigned short  u16x16;
typedef __attribute__((ext_vector_type(8)))  unsigned short  u16x8;
typedef __attribute__((ext_vector_type(4)))  unsigned short  u16x4;

__device__ __forceinline__ unsigned short f2bf(float f) {
  unsigned u = __builtin_bit_cast(unsigned, f);
  u += 0x7FFFu + ((u >> 16) & 1u);           // round-to-nearest-even
  return (unsigned short)(u >> 16);
}

// A-fragment (16x32 bf16) from LDS, per ISA 16-bit A layout:
//   lane<16 : elems0-7 = K0..7,  elems8-15 = K16..23   (row M = lane)
//   lane>=16: elems0-7 = K8..15, elems8-15 = K24..31   (row M = lane-16)
__device__ __forceinline__ v16bf load_a_frag(const unsigned short* A, int lda,
                                             int lane, int k0) {
  const int m  = lane & 15;
  const int kb = (lane & 16) ? 8 : 0;
  const unsigned short* p = A + m * lda + k0 + kb;
  u16x8 lo = *(const u16x8*)p;          // 16B aligned -> ds_load_b128
  u16x8 hi = *(const u16x8*)(p + 16);
  u16x16 t;
#pragma unroll
  for (int i = 0; i < 8; ++i) { t[i] = lo[i]; t[i + 8] = hi[i]; }
  return __builtin_bit_cast(v16bf, t);
}

// B-fragment from pre-swizzled weights: [kstep][ntile][lane][16] contiguous.
__device__ __forceinline__ v16bf load_b_frag(const unsigned short* Wsz,
                                             int ks, int nt, int lane) {
  return __builtin_bit_cast(
      v16bf, *(const u16x16*)(Wsz + (((size_t)ks * 8 + nt) * 32 + lane) * 16));
}

template <int KSTEPS, int LDA>
__device__ __forceinline__ void gemm_pair(const unsigned short* A,
                                          const unsigned short* Wsz,
                                          int lane, int nt0,
                                          v8f& acc0, v8f& acc1) {
#pragma unroll
  for (int k = 0; k < KSTEPS; ++k) {
    v16bf a  = load_a_frag(A, LDA, lane, k * 32);
    v16bf b0 = load_b_frag(Wsz, k, nt0, lane);
    acc0 = __builtin_amdgcn_wmma_f32_16x16x32_bf16(false, a, false, b0,
                                                   (short)0, acc0, false, false);
    v16bf b1 = load_b_frag(Wsz, k, nt0 + 1, lane);
    acc1 = __builtin_amdgcn_wmma_f32_16x16x32_bf16(false, a, false, b1,
                                                   (short)0, acc1, false, false);
  }
}

// Swizzle f32 row-major weights [fan_in,128] into per-lane B-fragment order.
// flat index = ((ks*8 + nt)*32 + lane)*16 + e ; K = ks*32 + (lane>=16?8:0)+(e&7)+(e>=8?16:0)
__global__ __launch_bounds__(256) void conv_weights_kernel(
    const float* __restrict__ W, unsigned short* __restrict__ out, int fan_in) {
  int i = blockIdx.x * 256 + threadIdx.x;
  if (i >= fan_in * 128) return;
  int e    = i & 15;
  int lane = (i >> 4) & 31;
  int nt   = (i >> 9) & 7;
  int ks   = i >> 12;
  int k    = ks * 32 + ((lane & 16) ? 8 : 0) + (e & 7) + ((e & 8) ? 16 : 0);
  int n    = nt * 16 + (lane & 15);
  out[i]   = f2bf(W[(size_t)k * 128 + n]);
}

// ---------------------------------------------------------------------------
// Edge stage: relu([e, x[row], x[col], u[g], face[f0], face[f1]] @ We + be)
// + scatter-add into sent/recv/fs/fr + per-graph pool ea.
// ---------------------------------------------------------------------------
__global__ __launch_bounds__(128) void edge_kernel(
    const float* __restrict__ x, const float* __restrict__ edge_attr,
    const float* __restrict__ u, const float* __restrict__ face,
    const unsigned short* __restrict__ Wsz, const float* __restrict__ bias,
    const int* __restrict__ row_idx, const int* __restrict__ col_idx,
    const int* __restrict__ f0_idx, const int* __restrict__ f1_idx,
    const int* __restrict__ edge_batch,
    float* __restrict__ edge_out, float* __restrict__ sent,
    float* __restrict__ recv, float* __restrict__ fs, float* __restrict__ fr,
    float* __restrict__ ea) {
  __shared__ unsigned short A[16 * 776];          // 16 rows x 768 bf16 (+8 pad)
  __shared__ int s_row[16], s_col[16], s_f0[16], s_f1[16], s_g[16];

  const int    tid   = threadIdx.x;
  const size_t ebase = (size_t)blockIdx.x * 16;

  if (tid < 16) {
    s_row[tid] = row_idx[ebase + tid];
    s_col[tid] = col_idx[ebase + tid];
    s_f0[tid]  = f0_idx[ebase + tid];
    s_f1[tid]  = f1_idx[ebase + tid];
    s_g[tid]   = edge_batch[ebase + tid];
  }
  __syncthreads();

  // Gather + f32->bf16: 16 rows * 6 segments * 32 float4.
#pragma unroll 4
  for (int i = tid; i < 16 * 6 * 32; i += 128) {
    int q = i & 31;
    int s = (i >> 5) % 6;
    int r = i / 192;
    const float* src;
    switch (s) {
      case 0:  src = edge_attr + (ebase + r) * 128; break;
      case 1:  src = x    + (size_t)s_row[r] * 128; break;
      case 2:  src = x    + (size_t)s_col[r] * 128; break;
      case 3:  src = u    + (size_t)s_g[r]   * 128; break;
      case 4:  src = face + (size_t)s_f0[r]  * 128; break;
      default: src = face + (size_t)s_f1[r]  * 128; break;
    }
    float4 v = ((const float4*)src)[q];
    u16x4 pk = {f2bf(v.x), f2bf(v.y), f2bf(v.z), f2bf(v.w)};
    *(u16x4*)(A + r * 776 + s * 128 + q * 4) = pk;   // ds_store_b64
  }
  __syncthreads();

  const int wave = tid >> 5, lane = tid & 31;
  const int nt0 = wave * 2;
  v8f acc0 = {}, acc1 = {};
  gemm_pair<24, 776>(A, Wsz, lane, nt0, acc0, acc1);

  const int n    = lane & 15;
  const int mofs = (lane & 16) ? 8 : 0;
#pragma unroll
  for (int t = 0; t < 2; ++t) {
    const v8f acc = t ? acc1 : acc0;
    const int col = (nt0 + t) * 16 + n;
    const float bv = bias[col];
    float psum = 0.f;
#pragma unroll
    for (int j = 0; j < 8; ++j) {
      const int m = j + mofs;
      float val = acc[j] + bv;
      val = fmaxf(val, 0.f);
      edge_out[(ebase + m) * 128 + col] = val;
      atomicAdd(&sent[(size_t)s_row[m] * 128 + col], val);
      atomicAdd(&recv[(size_t)s_col[m] * 128 + col], val);
      atomicAdd(&fs[(size_t)s_f0[m] * 128 + col], val);
      atomicAdd(&fr[(size_t)s_f1[m] * 128 + col], val);
      psum += val;
    }
    atomicAdd(&ea[(size_t)s_g[mofs] * 128 + col], psum);  // tile spans one graph
  }
}

// ---------------------------------------------------------------------------
// Generic 4-segment stage: relu([seg0,seg1,seg2,u[g]] @ W + b)
// MODE 0 = node (pool into `pool`), 1 = face (mask + pool), 2 = global (no pool)
// ---------------------------------------------------------------------------
template <int MODE>
__global__ __launch_bounds__(128) void stage4_kernel(
    const float* __restrict__ seg0, const float* __restrict__ seg1,
    const float* __restrict__ seg2, const float* __restrict__ u,
    const unsigned short* __restrict__ Wsz, const float* __restrict__ bias,
    const int* __restrict__ batch, const unsigned char* __restrict__ mask,
    float* __restrict__ out, float* __restrict__ pool) {
  __shared__ unsigned short A[16 * 520];          // 16 rows x 512 bf16 (+8 pad)
  __shared__ int s_g[16];
  __shared__ unsigned char s_mask[16];

  const int    tid  = threadIdx.x;
  const size_t base = (size_t)blockIdx.x * 16;

  if (tid < 16) {
    s_g[tid]    = (MODE == 2) ? 0 : batch[base + tid];
    s_mask[tid] = (MODE == 1) ? mask[base + tid] : (unsigned char)0;
  }
  __syncthreads();

#pragma unroll 4
  for (int i = tid; i < 16 * 4 * 32; i += 128) {
    int q = i & 31;
    int s = (i >> 5) & 3;
    int r = i >> 7;
    const float* src;
    switch (s) {
      case 0:  src = seg0 + (base + r) * 128; break;
      case 1:  src = seg1 + (base + r) * 128; break;
      case 2:  src = seg2 + (base + r) * 128; break;
      default: src = u + (size_t)((MODE == 2) ? (int)(base + r) : s_g[r]) * 128;
    }
    float4 v = ((const float4*)src)[q];
    u16x4 pk = {f2bf(v.x), f2bf(v.y), f2bf(v.z), f2bf(v.w)};
    *(u16x4*)(A + r * 520 + s * 128 + q * 4) = pk;
  }
  __syncthreads();

  const int wave = tid >> 5, lane = tid & 31;
  const int nt0 = wave * 2;
  v8f acc0 = {}, acc1 = {};
  gemm_pair<16, 520>(A, Wsz, lane, nt0, acc0, acc1);

  const int n    = lane & 15;
  const int mofs = (lane & 16) ? 8 : 0;
#pragma unroll
  for (int t = 0; t < 2; ++t) {
    const v8f acc = t ? acc1 : acc0;
    const int col = (nt0 + t) * 16 + n;
    const float bv = bias[col];
    float psum = 0.f;
#pragma unroll
    for (int j = 0; j < 8; ++j) {
      const int m = j + mofs;
      float val = fmaxf(acc[j] + bv, 0.f);
      if (MODE == 1 && s_mask[m]) val = 0.f;      // face_mask -> zero
      out[(base + m) * 128 + col] = val;
      psum += val;
    }
    if (MODE != 2) atomicAdd(&pool[(size_t)s_g[mofs] * 128 + col], psum);
  }
}

// ---------------------------------------------------------------------------
extern "C" void kernel_launch(void* const* d_in, const int* in_sizes, int n_in,
                              void* d_out, int out_size, void* d_ws,
                              size_t ws_size, hipStream_t stream) {
  const float* x         = (const float*)d_in[0];
  const float* edge_attr = (const float*)d_in[1];
  const float* u         = (const float*)d_in[2];
  const float* face      = (const float*)d_in[3];
  const float* W_edge    = (const float*)d_in[4];
  const float* b_edge    = (const float*)d_in[5];
  const float* W_node    = (const float*)d_in[6];
  const float* b_node    = (const float*)d_in[7];
  const float* W_face    = (const float*)d_in[8];
  const float* b_face    = (const float*)d_in[9];
  const float* W_glob    = (const float*)d_in[10];
  const float* b_glob    = (const float*)d_in[11];
  const int*   edge_index = (const int*)d_in[12];
  const int*   face_index = (const int*)d_in[13];
  const int*   node_batch = (const int*)d_in[14];
  const int*   edge_batch = (const int*)d_in[15];
  const int*   face_batch = (const int*)d_in[16];
  const unsigned char* face_mask = (const unsigned char*)d_in[17];

  const int N = in_sizes[0] / 128;
  const int E = in_sizes[1] / 128;
  const int B = in_sizes[2] / 128;
  const int F = in_sizes[3] / 128;

  const int* row = edge_index;
  const int* col = edge_index + E;
  const int* fi0 = face_index;
  const int* fi1 = face_index + E;

  // Output layout: x' | edge' | u' | face'
  float* out      = (float*)d_out;
  float* x_out    = out;
  float* e_out    = x_out + (size_t)N * 128;
  float* u_out    = e_out + (size_t)E * 128;
  float* f_out    = u_out + (size_t)B * 128;

  // Workspace carve-up.
  char* p = (char*)d_ws;
  float* sent = (float*)p;  p += (size_t)N * 128 * 4;
  float* recv = (float*)p;  p += (size_t)N * 128 * 4;
  float* fs   = (float*)p;  p += (size_t)F * 128 * 4;
  float* fr   = (float*)p;  p += (size_t)F * 128 * 4;
  float* na   = (float*)p;  p += (size_t)B * 128 * 4;
  float* ea   = (float*)p;  p += (size_t)B * 128 * 4;
  float* fa   = (float*)p;  p += (size_t)B * 128 * 4;
  unsigned short* We = (unsigned short*)p;  p += (size_t)768 * 128 * 2;
  unsigned short* Wn = (unsigned short*)p;  p += (size_t)512 * 128 * 2;
  unsigned short* Wf = (unsigned short*)p;  p += (size_t)512 * 128 * 2;
  unsigned short* Wg = (unsigned short*)p;  p += (size_t)512 * 128 * 2;

  const size_t accum_bytes = ((size_t)2 * N + 2 * F + 3 * B) * 128 * 4;
  hipMemsetAsync(d_ws, 0, accum_bytes, stream);

  conv_weights_kernel<<<(768 * 128 + 255) / 256, 256, 0, stream>>>(W_edge, We, 768);
  conv_weights_kernel<<<(512 * 128 + 255) / 256, 256, 0, stream>>>(W_node, Wn, 512);
  conv_weights_kernel<<<(512 * 128 + 255) / 256, 256, 0, stream>>>(W_face, Wf, 512);
  conv_weights_kernel<<<(512 * 128 + 255) / 256, 256, 0, stream>>>(W_glob, Wg, 512);

  edge_kernel<<<E / 16, 128, 0, stream>>>(x, edge_attr, u, face, We, b_edge,
                                          row, col, fi0, fi1, edge_batch,
                                          e_out, sent, recv, fs, fr, ea);

  stage4_kernel<0><<<N / 16, 128, 0, stream>>>(x, sent, recv, u, Wn, b_node,
                                               node_batch, nullptr, x_out, na);
  stage4_kernel<1><<<F / 16, 128, 0, stream>>>(face, fs, fr, u, Wf, b_face,
                                               face_batch, face_mask, f_out, fa);
  stage4_kernel<2><<<B / 16, 128, 0, stream>>>(u, na, ea, u, Wg, b_glob,
                                               nullptr, nullptr, u_out, nullptr);
}
